// ConvolutionLayer_712964571640
// MI455X (gfx1250) — matrix operand order repaired and checked by reading, not compiled, not run on hardware
//
#include <hip/hip_runtime.h>
#include <math.h>

typedef __attribute__((ext_vector_type(16))) _Float16 v16h;
typedef __attribute__((ext_vector_type(8)))  _Float16 v8h;
typedef __attribute__((ext_vector_type(8)))  float    v8f;

#define T_LEN   65536
#define B_SZ    4
#define TT      256            // time steps per block (2 subtiles of 16 per wave)
#define XT      (TT + 16)      // with causal halo (dilation 8, K=3 -> 16)
#define XPITCH  80             // halves per xs row: 160B rows, 32B-aligned, bank-spread

// LDS layout (bytes)
#define OFF_WCONV 0            // 3*128*64 f16 = 49152
#define OFF_WOUT  49152        // 64*64 f16   = 8192
#define OFF_BCONV 57344        // 128 f32     = 512
#define OFF_BOUT  57856        // 64 f32      = 256
#define OFF_XS    58112        // 272*80 f16  = 43520
#define OFF_GS    101632       // 8*16*64 f16 = 16384
#define SMEM_BYTES 118016

__device__ __forceinline__ v16h cat8(v8h lo, v8h hi8) {
  return __builtin_shufflevector(lo, hi8, 0,1,2,3,4,5,6,7,8,9,10,11,12,13,14,15);
}

__device__ __forceinline__ float fast_tanh(float x) {
#if __has_builtin(__builtin_amdgcn_tanhf)
  return __builtin_amdgcn_tanhf(x);     // hardware V_TANH_F32 (CDNA5 trans op)
#else
  return tanhf(x);
#endif
}

__device__ __forceinline__ float fast_rcp(float x) {
#if __has_builtin(__builtin_amdgcn_rcpf)
  return __builtin_amdgcn_rcpf(x);      // V_RCP_F32
#else
  return 1.0f / x;
#endif
}

extern "C" __global__ void __launch_bounds__(256, 1)
wavenet_fused_kernel(const float* __restrict__ x,
                     const float* __restrict__ wconv,
                     const float* __restrict__ bconv,
                     const float* __restrict__ wout,
                     const float* __restrict__ bout,
                     float* __restrict__ out)
{
  extern __shared__ char smem[];
  _Float16* ws_conv = (_Float16*)(smem + OFF_WCONV);   // [k][o][i]
  _Float16* ws_out  = (_Float16*)(smem + OFF_WOUT);    // [o][i]
  float*    bcl     = (float*)   (smem + OFF_BCONV);
  float*    bol     = (float*)   (smem + OFF_BOUT);
  _Float16* xs      = (_Float16*)(smem + OFF_XS);      // [tl][ch], pitch XPITCH
  _Float16* gs      = (_Float16*)(smem + OFF_GS);      // per-wave [16][64]

  const int tid = threadIdx.x;
  const int b   = blockIdx.y;
  const int t0  = blockIdx.x * TT;

  // ---- stage conv weights: wconv[o][i][k] -> ws_conv[k][o][i] (f16) ----
  for (int e = tid; e < 3 * 128 * 64; e += 256) {
    int k = e >> 13;           // / 8192
    int r = e & 8191;
    int o = r >> 6, i = r & 63;
    ws_conv[e] = (_Float16)wconv[(o * 64 + i) * 3 + k];
  }
  // ---- stage 1x1 weights (already [o][i]) ----
  for (int e = tid; e < 64 * 64; e += 256)
    ws_out[e] = (_Float16)wout[e];
  if (tid < 128) bcl[tid] = bconv[tid];
  if (tid < 64)  bol[tid] = bout[tid];

  // ---- stage x tile transposed to [t][ch] f16, zero left halo ----
  const float* xbp = x + (size_t)b * 64 * T_LEN;
  for (int e = tid; e < 64 * XT; e += 256) {
    int ch = e / XT;
    int tl = e - ch * XT;
    int tg = t0 - 16 + tl;
    float v = (tg >= 0) ? xbp[ch * T_LEN + tg] : 0.0f;
    xs[tl * XPITCH + ch] = (_Float16)v;
  }
  __syncthreads();

  const int lane = tid & 31;
  const int ln   = lane & 15;   // N (time) within 16x16 tile
  const int hi   = lane >> 4;   // lane-half selector
  const int wv   = tid >> 5;

  const size_t bbase = (size_t)b * 64 * T_LEN;
  float* skip = out + (size_t)B_SZ * 64 * T_LEN;
  _Float16* gsl = &gs[wv * 16 * 64];

  // A-fragment LDS half-offset for flattened conv index i = m*6 + k*2 + ks
  auto aoff_conv = [&](int i) {
    int m = i / 6, r = i - m * 6, k = r >> 1, ks = r & 1;
    return ((k * 128 + m * 16 + ln) * 64) + ks * 32 + hi * 8;
  };
  auto aoff_out = [&](int i) {            // i = m*2 + ks
    int m = i >> 1, ks = i & 1;
    return ((m * 16 + ln) * 64) + ks * 32 + hi * 8;
  };

  for (int nt = 0; nt < 2; ++nt) {
    const int tb = nt * 128 + wv * 16;   // this wave's time base within block tile

    // ---- B fragments for the 3 taps x 2 K-steps (reused across 8 M-tiles) ----
    v16h xbf[6];
#pragma unroll
    for (int k = 0; k < 3; ++k)
#pragma unroll
      for (int ks = 0; ks < 2; ++ks)
        xbf[k * 2 + ks] =
            cat8(*(const v8h*)&xs[(tb + ln + 8 * k) * XPITCH + ks * 32 + hi * 8],
                 *(const v8h*)&xs[(tb + ln + 8 * k) * XPITCH + ks * 32 + 16 + hi * 8]);

    // ---- H accumulators (128 channels = 8 M-tiles), init with conv bias ----
    v8f hacc[8];
#pragma unroll
    for (int m = 0; m < 8; ++m)
      hacc[m] = *(const v8f*)&bcl[m * 16 + hi * 8];

    // ---- dilated conv: 48 WMMAs, A-fragments software-pipelined (depth 2) ----
    v16h a0, a1;
    { const v8h* p = (const v8h*)&ws_conv[aoff_conv(0)]; a0 = cat8(p[0], p[2]); }
    { const v8h* p = (const v8h*)&ws_conv[aoff_conv(1)]; a1 = cat8(p[0], p[2]); }
#pragma unroll
    for (int i = 0; i < 48; ++i) {
      v16h a_use = a0;
      a0 = a1;
      if (i + 2 < 48) {
        const v8h* p = (const v8h*)&ws_conv[aoff_conv(i + 2)];
        a1 = cat8(p[0], p[2]);
      }
      int m = i / 6, r = i - m * 6;
      hacc[m] = __builtin_amdgcn_wmma_f32_16x16x32_f16(
          false, a_use, false, xbf[r], (short)0, hacc[m], false, false);
    }

    // ---- gated activation (hardware tanh/exp/rcp); stash gated f16 in LDS ----
    float gt[4][8];
#pragma unroll
    for (int mp = 0; mp < 4; ++mp) {
      v8h gh;
#pragma unroll
      for (int v = 0; v < 8; ++v) {
        float ta = hacc[mp][v];
        float sa = hacc[mp + 4][v];
        float g  = fast_tanh(ta) * fast_rcp(1.0f + __expf(-sa));
        gt[mp][v] = g;
        gh[v] = (_Float16)g;
      }
      *(v8h*)&gsl[ln * 64 + mp * 16 + hi * 8] = gh;
    }

    // ---- store skip = gated ----
    const int tg = t0 + tb + ln;
#pragma unroll
    for (int mp = 0; mp < 4; ++mp)
#pragma unroll
      for (int v = 0; v < 8; ++v) {
        int c = mp * 16 + hi * 8 + v;
        skip[bbase + (size_t)c * T_LEN + tg] = gt[mp][v];
      }

    // ---- 1x1 output conv: 8 WMMAs (gated re-shaped D->B via per-wave LDS) ----
    v16h gbf[2];
#pragma unroll
    for (int ks = 0; ks < 2; ++ks)
      gbf[ks] = cat8(*(const v8h*)&gsl[ln * 64 + ks * 32 + hi * 8],
                     *(const v8h*)&gsl[ln * 64 + ks * 32 + 16 + hi * 8]);

    v8f oacc[4];
#pragma unroll
    for (int m = 0; m < 4; ++m)
      oacc[m] = *(const v8f*)&bol[m * 16 + hi * 8];

    v16h c0, c1;
    { const v8h* p = (const v8h*)&ws_out[aoff_out(0)]; c0 = cat8(p[0], p[2]); }
    { const v8h* p = (const v8h*)&ws_out[aoff_out(1)]; c1 = cat8(p[0], p[2]); }
#pragma unroll
    for (int i = 0; i < 8; ++i) {
      v16h a_use = c0;
      c0 = c1;
      if (i + 2 < 8) {
        const v8h* p = (const v8h*)&ws_out[aoff_out(i + 2)];
        c1 = cat8(p[0], p[2]);
      }
      int m = i >> 1, ks = i & 1;
      oacc[m] = __builtin_amdgcn_wmma_f32_16x16x32_f16(
          false, a_use, false, gbf[ks], (short)0, oacc[m], false, false);
    }
#pragma unroll
    for (int m = 0; m < 4; ++m)
#pragma unroll
      for (int v = 0; v < 8; ++v) {
        int o = m * 16 + hi * 8 + v;
        out[bbase + (size_t)o * T_LEN + tg] = oacc[m][v];
      }
  }
}

extern "C" void kernel_launch(void* const* d_in, const int* in_sizes, int n_in,
                              void* d_out, int out_size, void* d_ws, size_t ws_size,
                              hipStream_t stream) {
  (void)in_sizes; (void)n_in; (void)out_size; (void)d_ws; (void)ws_size;
  const float* x     = (const float*)d_in[0];
  const float* wconv = (const float*)d_in[1];
  const float* bconv = (const float*)d_in[2];
  const float* wout  = (const float*)d_in[3];
  const float* bout  = (const float*)d_in[4];
  float* out = (float*)d_out;

  dim3 grid(T_LEN / TT, B_SZ);   // 256 x 4 blocks
  dim3 block(256);               // 8 wave32
  wavenet_fused_kernel<<<grid, block, SMEM_BYTES, stream>>>(
      x, wconv, bconv, wout, bout, out);
}
